// WordGenerator_9852654977108
// MI455X (gfx1250) — matrix operand order repaired and checked by reading, not compiled, not run on hardware
//
#include <hip/hip_runtime.h>
#include <hip/hip_bf16.h>

typedef __bf16 bf16;
typedef __attribute__((ext_vector_type(8)))  __bf16 v8bf;
typedef __attribute__((ext_vector_type(16))) __bf16 v16bf;
typedef __attribute__((ext_vector_type(8)))  float  v8f;

static constexpr int Vv = 32000;
static constexpr int Hh = 512;
static constexpr int Bb = 256;
static constexpr int Ss = 128;
static constexpr int Ll = 5;

// ---------------- prep kernels ----------------

__global__ void f32_to_bf16_kernel(const float* __restrict__ src,
                                   bf16* __restrict__ dst, long n) {
  long i = (long)blockIdx.x * blockDim.x + threadIdx.x;
  long stride = (long)gridDim.x * blockDim.x;
  for (; i < n; i += stride) dst[i] = (bf16)src[i];
}

__global__ void bias_sum_kernel(const float* __restrict__ a,
                                const float* __restrict__ b,
                                float* __restrict__ o, int n) {
  int i = blockIdx.x * blockDim.x + threadIdx.x;
  if (i < n) o[i] = a[i] + b[i];
}

// embedding gather for one timestep, reproducing torch .view(S,B,H) on (B,S,H)
__global__ void embed_kernel(const int* __restrict__ x,
                             const float* __restrict__ emb,
                             bf16* __restrict__ xbf, int t) {
  int idx = blockIdx.x * blockDim.x + threadIdx.x;   // over B*H
  int b = idx >> 9;          // H = 512
  int h = idx & (Hh - 1);
  int flat = t * Bb + b;
  int tok = x[(flat >> 7) * Ss + (flat & (Ss - 1))]; // S = 128
  xbf[idx] = (bf16)emb[(long)tok * Hh + h];
}

// ---------------- LSTM cell elementwise ----------------

__global__ void lstm_cell_kernel(const float* __restrict__ gates,
                                 float* __restrict__ c,
                                 bf16* __restrict__ h) {
  int idx = blockIdx.x * blockDim.x + threadIdx.x;   // over B*H
  int b = idx >> 9;
  int j = idx & (Hh - 1);
  const float* gr = gates + (long)b * (4 * Hh);
  float ig = gr[j];
  float fg = gr[Hh + j];
  float gg = gr[2 * Hh + j];
  float og = gr[3 * Hh + j];
  float i = 1.f / (1.f + __expf(-ig));
  float f = 1.f / (1.f + __expf(-fg));
  float g = tanhf(gg);
  float o = 1.f / (1.f + __expf(-og));
  float cn = f * c[idx] + i * g;
  float hn = o * tanhf(cn);
  c[idx] = cn;
  h[idx] = (bf16)hn;
}

// ---------------- bf16 WMMA GEMM: C[M,N] = A0[M,K]*W0[N,K]^T (+ A1*W1^T) + bias ----
// Per-wave 32x32 register block (2x2 tiles of 16x16); workgroup = 8 waves (2M x 4N)
// -> 64x128 block tile. Requires M % 64 == 0, N % 128 == 0, K % 32 == 0.

__device__ __forceinline__ v16bf load_a_frag(const bf16* __restrict__ p, int half) {
  // A 16x32 bf16 fragment: lane half 0 holds K [0..8)+[16..24),
  // half 1 holds K [8..16)+[24..32) of this 32-wide K slice.
  v8bf a0 = *(const v8bf*)(p + half * 8);
  v8bf a1 = *(const v8bf*)(p + 16 + half * 8);
  return __builtin_shufflevector(a0, a1, 0, 1, 2, 3, 4, 5, 6, 7,
                                 8, 9, 10, 11, 12, 13, 14, 15);
}

__device__ __forceinline__ void wmma_block2x2(
    const bf16* __restrict__ A, const bf16* __restrict__ W, int K,
    int mBase, int nBase, int r, int half,
    v8f& c00, v8f& c01, v8f& c10, v8f& c11) {
  const bf16* Ar0 = A + (long)(mBase + r) * K;
  const bf16* Ar1 = A + (long)(mBase + 16 + r) * K;
  const bf16* Wr0 = W + (long)(nBase + r) * K;
  const bf16* Wr1 = W + (long)(nBase + 16 + r) * K;
#pragma unroll 2
  for (int kb = 0; kb < K; kb += 32) {
    v16bf a0 = load_a_frag(Ar0 + kb, half);
    v16bf a1 = load_a_frag(Ar1 + kb, half);
    v16bf b0 = load_a_frag(Wr0 + kb, half);  // B 32x16: same K split as A (8h+[0..8), 16+8h+[0..8))
    v16bf b1 = load_a_frag(Wr1 + kb, half);
    c00 = __builtin_amdgcn_wmma_f32_16x16x32_bf16(false, a0, false, b0, (short)0, c00, false, false);
    c01 = __builtin_amdgcn_wmma_f32_16x16x32_bf16(false, a0, false, b1, (short)0, c01, false, false);
    c10 = __builtin_amdgcn_wmma_f32_16x16x32_bf16(false, a1, false, b0, (short)0, c10, false, false);
    c11 = __builtin_amdgcn_wmma_f32_16x16x32_bf16(false, a1, false, b1, (short)0, c11, false, false);
  }
}

__device__ __forceinline__ void store_tile(float* __restrict__ C, long N,
                                           int m, int n, int r, int half,
                                           v8f acc, float bv) {
  // C/D layout: VGPR v of lane (r,half) holds D[mTile + v + half*8][nTile + r]
  float* Cout = C + (long)(m + half * 8) * N + n + r;
#pragma unroll
  for (int v = 0; v < 8; ++v) Cout[(long)v * N] = acc[v] + bv;
}

__global__ __launch_bounds__(256) void gemm_wmma_kernel(
    const bf16* __restrict__ A0, const bf16* __restrict__ W0,
    const bf16* __restrict__ A1, const bf16* __restrict__ W1,
    const float* __restrict__ bias, float* __restrict__ C,
    int N, int K) {
  const int lane = threadIdx.x & 31;
  const int wave = threadIdx.x >> 5;      // 8 waves: 2 (M) x 4 (N)
  const int half = lane >> 4;
  const int r    = lane & 15;
  const int mBase = blockIdx.y * 64 + (wave & 1) * 32;
  const int nBase = blockIdx.x * 128 + (wave >> 1) * 32;

  v8f c00 = {}, c01 = {}, c10 = {}, c11 = {};
  wmma_block2x2(A0, W0, K, mBase, nBase, r, half, c00, c01, c10, c11);
  if (A1) {
    wmma_block2x2(A1, W1, K, mBase, nBase, r, half, c00, c01, c10, c11);
  }

  const float bv0 = bias[nBase + r];
  const float bv1 = bias[nBase + 16 + r];
  store_tile(C, N, mBase,      nBase,      r, half, c00, bv0);
  store_tile(C, N, mBase,      nBase + 16, r, half, c01, bv1);
  store_tile(C, N, mBase + 16, nBase,      r, half, c10, bv0);
  store_tile(C, N, mBase + 16, nBase + 16, r, half, c11, bv1);
}

// ---------------- host side ----------------

extern "C" void kernel_launch(void* const* d_in, const int* in_sizes, int n_in,
                              void* d_out, int out_size, void* d_ws, size_t ws_size,
                              hipStream_t stream) {
  (void)in_sizes; (void)n_in; (void)out_size; (void)ws_size;

  const int*   x    = (const int*)d_in[0];
  const float* h0   = (const float*)d_in[1];
  const float* c0   = (const float*)d_in[2];
  const float* emb  = (const float*)d_in[3];
  const float* Wih  = (const float*)d_in[4];
  const float* Whh  = (const float*)d_in[5];
  const float* bih  = (const float*)d_in[6];
  const float* bhh  = (const float*)d_in[7];
  const float* Wout = (const float*)d_in[8];
  const float* bout = (const float*)d_in[9];
  float* out = (float*)d_out;

  // workspace layout (bytes), all 32B-aligned for v16bf loads
  char* ws = (char*)d_ws;
  bf16*  WihB  = (bf16*)(ws + 0);          // L*4H*H bf16 = 10,485,760 B
  bf16*  WhhB  = (bf16*)(ws + 10485760);   // 10,485,760 B
  bf16*  WoutB = (bf16*)(ws + 20971520);   // V*H bf16   = 32,768,000 B
  float* bias  = (float*)(ws + 53739520);  // L*4H f32   =     40,960 B
  bf16*  hB    = (bf16*)(ws + 53780480);   // L*B*H bf16 =  1,310,720 B
  float* cS    = (float*)(ws + 55091200);  // L*B*H f32  =  2,621,440 B
  bf16*  xB    = (bf16*)(ws + 57712640);   // B*H bf16   =    262,144 B
  float* gates = (float*)(ws + 57974784);  // B*4H f32   =  2,097,152 B
  // total ~60.1 MB

  const long nW = (long)Ll * 4 * Hh * Hh;  // 5,242,880 per weight set

  f32_to_bf16_kernel<<<2048, 256, 0, stream>>>(Wih, WihB, nW);
  f32_to_bf16_kernel<<<2048, 256, 0, stream>>>(Whh, WhhB, nW);
  f32_to_bf16_kernel<<<4096, 256, 0, stream>>>(Wout, WoutB, (long)Vv * Hh);
  f32_to_bf16_kernel<<<512,  256, 0, stream>>>(h0, hB, (long)Ll * Bb * Hh);
  bias_sum_kernel<<<(Ll * 4 * Hh + 255) / 256, 256, 0, stream>>>(
      bih, bhh, bias, Ll * 4 * Hh);
  hipMemcpyAsync(cS, c0, (size_t)Ll * Bb * Hh * sizeof(float),
                 hipMemcpyDeviceToDevice, stream);

  const dim3 gGates(4 * Hh / 128, Bb / 64);  // (16, 4) = 64 blocks
  const dim3 gElem(Bb * Hh / 256);           // 512 blocks

  for (int t = 0; t < Ss; ++t) {
    embed_kernel<<<gElem, 256, 0, stream>>>(x, emb, xB, t);
    for (int l = 0; l < Ll; ++l) {
      const bf16* A0 = (l == 0) ? xB : hB + (long)(l - 1) * Bb * Hh;
      gemm_wmma_kernel<<<gGates, 256, 0, stream>>>(
          A0, WihB + (long)l * 4 * Hh * Hh,
          hB + (long)l * Bb * Hh, WhhB + (long)l * 4 * Hh * Hh,
          bias + l * 4 * Hh, gates, 4 * Hh, Hh);
      lstm_cell_kernel<<<gElem, 256, 0, stream>>>(
          gates, cS + (long)l * Bb * Hh, hB + (long)l * Bb * Hh);
    }
  }

  const dim3 gOut(Vv / 128, Bb / 64);        // (250, 4) = 1000 blocks
  gemm_wmma_kernel<<<gOut, 256, 0, stream>>>(
      hB + (long)(Ll - 1) * Bb * Hh, WoutB,
      nullptr, nullptr, bout, out, Vv, Hh);
}
